// GATNetwork_46059229283058
// MI455X (gfx1250) — matrix operand order repaired and checked by reading, not compiled, not run on hardware
//
#include <hip/hip_runtime.h>
#include <hip/hip_bf16.h>

// ---------------------------------------------------------------------------
// GATv2 (H=4, C=32) + global-max-pool + dueling head on gfx1250.
//   - node transforms: bf16 WMMA GEMM, A-tile staged to LDS via TDM
//     (tensor_load_to_lds + s_wait_tensorcnt), padded LDS layout for
//     conflict-free ds_load_b128 fragment reads.
//   - edge softmax/aggregate: L2-resident atomics (~92MB < 192MB L2).
// ---------------------------------------------------------------------------

#define HH 4
#define CC 32
#define HC 128   // H*C == D_IN == HID
#define GG 64
#define NEG_SLOPE 0.2f

typedef __bf16 bf16;
typedef __attribute__((ext_vector_type(16))) __bf16 v16bf;
typedef __attribute__((ext_vector_type(8)))  __bf16 v8bf;
typedef __attribute__((ext_vector_type(8)))  float  v8f;
typedef __attribute__((ext_vector_type(4)))  unsigned int u32x4;
typedef __attribute__((ext_vector_type(8)))  int i32x8;
typedef __attribute__((ext_vector_type(4)))  int i32x4;

// LDS A-tile: 64 rows, stride 136 bf16 (272B) -> bank-conflict-free b128 reads
#define LDS_STRIDE 136

// ---- ordered-float <-> uint mapping so atomicMax(u32) == float max --------
__device__ __forceinline__ unsigned ford(float f) {
    unsigned u = __float_as_uint(f);
    return (u >> 31) ? ~u : (u | 0x80000000u);
}
__device__ __forceinline__ float dford(unsigned u) {
    return __uint_as_float((u >> 31) ? (u ^ 0x80000000u) : ~u);
}
#define ORD_NEG_INF 0x007FFFFFu   // ford(-inf)

// ---------------------------------------------------------------------------
__global__ __launch_bounds__(256) void fill_u32(unsigned* p, unsigned v, int n) {
    int i = blockIdx.x * 256 + threadIdx.x;
    if (i < n) p[i] = v;
}
__global__ __launch_bounds__(256) void fill_f32(float* p, float v, int n) {
    int i = blockIdx.x * 256 + threadIdx.x;
    if (i < n) p[i] = v;
}

// x (f32) -> bf16
__global__ __launch_bounds__(256) void cvt_bf16(const float* __restrict__ in,
                                                bf16* __restrict__ out, int n) {
    int i = blockIdx.x * 256 + threadIdx.x;
    if (i < n) out[i] = (bf16)in[i];
}

// W[k][n] (128x128 f32) -> Wt[n][k] (bf16), so WMMA B loads are contiguous.
__global__ __launch_bounds__(256) void w_transpose_bf16(const float* __restrict__ W,
                                                        bf16* __restrict__ Wt) {
    int i = blockIdx.x * 256 + threadIdx.x;  // 16384 total
    int k = i >> 7, n = i & 127;
    Wt[n * 128 + k] = (bf16)W[k * 128 + n];
}

// ---------------------------------------------------------------------------
// out[64x128 tile] = A_bf16 @ W (Wt[n][k] bf16) + bias[n]
// Block = 256 thr = 8 waves = 8 N-tiles of 16; each wave does 4 M-tiles.
// A tile (64x128 bf16 = 16KB) staged into LDS by the Tensor Data Mover with
// 16B/row padding (descriptor pad_interval=64dw, pad_amount=4dw).
// ---------------------------------------------------------------------------
__global__ __launch_bounds__(256)
void gemm128_wmma(const bf16* __restrict__ A, const bf16* __restrict__ Bt,
                  const float* __restrict__ bias, float* __restrict__ C,
                  int Nrows) {
    __shared__ bf16 smem[64 * LDS_STRIDE];

    const int wave = threadIdx.x >> 5;
    const int lane = threadIdx.x & 31;
    const int m0 = blockIdx.x * 64;
    const int n0 = wave * 16;
    const int lr = lane & 15;
    const int hi = lane >> 4;           // 0: lanes 0-15, 1: lanes 16-31
    const int kA = hi * 8;              // A-frag K half offset (interleaved)
    const int kB = hi * 8;              // B-frag K half offset (interleaved)

#if __has_builtin(__builtin_amdgcn_tensor_load_to_lds) && \
    __has_builtin(__builtin_amdgcn_s_wait_tensorcnt)
    if (wave == 0) {
        // ---- build Tensor DMA Descriptor (D#), 2D tile 128 x 64 rows ----
        unsigned long long ga =
            (unsigned long long)(const void*)A + (unsigned long long)m0 * 256ull;
        unsigned lds_base = (unsigned)(size_t)(void*)smem;
        u32x4 g0;
        g0[0] = 1u;                                      // count=1 (valid), user mode
        g0[1] = lds_base;                                // lds_addr (bytes)
        g0[2] = (unsigned)(ga & 0xFFFFFFFFull);          // global_addr[31:0]
        g0[3] = (unsigned)((ga >> 32) & 0x01FFFFFFull)   // global_addr[56:32]
              | (2u << 30);                              // type = 2 ("image")
        i32x8 g1;
        g1[0] = (int)0x07510000u;   // data_size=2B | pad_en | interval=64dw | amount=4dw
        g1[1] = (int)(128u << 16);                       // tensor_dim0[15:0] = 128
        g1[2] = (int)(((unsigned)Nrows & 0xFFFFu) << 16);// dim0 hi=0 | tensor_dim1 lo
        g1[3] = (int)((((unsigned)Nrows >> 16) & 0xFFFFu)// tensor_dim1 hi
              | (128u << 16));                           // tile_dim0 = 128
        g1[4] = 64;                                      // tile_dim1=64, tile_dim2=0
        g1[5] = 128;                                     // tensor_dim0_stride = 128
        g1[6] = 0;
        g1[7] = 0;
        i32x4 gz4 = {0, 0, 0, 0};
        i32x8 gz8 = {0, 0, 0, 0, 0, 0, 0, 0};
        __builtin_amdgcn_tensor_load_to_lds(g0, g1, gz4, gz4, gz8, 0);
        __builtin_amdgcn_s_wait_tensorcnt(0);
    }
    __syncthreads();
#else
    // fallback: cooperative staged copy with identical padded layout
    for (int i = threadIdx.x; i < 64 * 16; i += 256) {
        int row = i >> 4, seg = i & 15;
        *(v8bf*)&smem[row * LDS_STRIDE + seg * 8] =
            *(const v8bf*)(A + (size_t)(m0 + row) * 128 + seg * 8);
    }
    __syncthreads();
#endif

    // ---- preload B fragments for all 4 K-steps (reused by 4 M-tiles) ----
    const bf16* bptr = Bt + (size_t)(n0 + lr) * 128 + kB;
    v8bf b0[4], b1[4];
    #pragma unroll
    for (int s = 0; s < 4; ++s) {
        b0[s] = *(const v8bf*)(bptr + 32 * s);
        b1[s] = *(const v8bf*)(bptr + 32 * s + 16);
    }

    const float bv = bias[n0 + lr];     // C layout: N = lane&15 for all 8 VGPRs

    #pragma unroll
    for (int mt = 0; mt < 4; ++mt) {
        const bf16* arow = &smem[(mt * 16 + lr) * LDS_STRIDE + kA];
        v8f c;
        #pragma unroll
        for (int i = 0; i < 8; ++i) c[i] = bv;

        #pragma unroll
        for (int s = 0; s < 4; ++s) {
            v8bf a0 = *(const v8bf*)(arow + 32 * s);        // K = kA .. +7
            v8bf a1 = *(const v8bf*)(arow + 32 * s + 16);   // K = kA+16 .. +23
            v16bf a, b;
            #pragma unroll
            for (int i = 0; i < 8; ++i) {
                a[i] = a0[i]; a[8 + i] = a1[i];
                b[i] = b0[s][i]; b[8 + i] = b1[s][i];
            }
            c = __builtin_amdgcn_wmma_f32_16x16x32_bf16(
                    false, a, false, b, (short)0, c, false, false);
        }

        // D layout: VGPR v -> M = v + 8*hi, N = lane&15. Grid covers full
        // 64-row tiles only -> no store guard needed.
        const size_t rbase = (size_t)(m0 + mt * 16 + 8 * hi);
        #pragma unroll
        for (int v = 0; v < 8; ++v)
            C[(rbase + v) * 128 + n0 + lr] = c[v];
    }
}

// scalar tail for rows not covered by full 64-row WMMA tiles (f32 exact)
__global__ __launch_bounds__(256)
void gemm_tail(const float* __restrict__ x, const float* __restrict__ W,
               const float* __restrict__ bias, float* __restrict__ C,
               int startRow, int M) {
    int i = blockIdx.x * 256 + threadIdx.x;
    int total = (M - startRow) * HC;
    if (i >= total) return;
    int row = startRow + (i >> 7), n = i & 127;
    float acc = bias[n];
    #pragma unroll 8
    for (int k = 0; k < HC; ++k) acc += x[(size_t)row * HC + k] * W[k * HC + n];
    C[(size_t)row * HC + n] = acc;
}

// ---------------------------------------------------------------------------
// Edge phase. Thread = (edge_with_self_loop, head). Et = E + N.
// ---------------------------------------------------------------------------
__device__ __forceinline__ void edge_ends(const int* __restrict__ ei, int E,
                                          int et, int& s, int& d) {
    if (et < E) { s = ei[et]; d = ei[E + et]; }
    else        { s = d = et - E; }
}

__global__ __launch_bounds__(256)
void edge_score(const float* __restrict__ xl, const float* __restrict__ xr,
                const float* __restrict__ att, const int* __restrict__ ei,
                int E, int N, float* __restrict__ ebuf, unsigned* __restrict__ emax) {
    int idx = blockIdx.x * 256 + threadIdx.x;
    int Et = E + N;
    if (idx >= Et * HH) return;
    int et = idx >> 2, h = idx & 3;
    int s, d; edge_ends(ei, E, et, s, d);

    const float4* pl = (const float4*)(xl + (size_t)s * HC + h * CC);
    const float4* pr = (const float4*)(xr + (size_t)d * HC + h * CC);
    const float4* pa = (const float4*)(att + h * CC);
    float acc = 0.f;
    #pragma unroll
    for (int q = 0; q < 8; ++q) {
        float4 a = pl[q], b = pr[q], w = pa[q];
        float m;
        m = a.x + b.x; m = (m > 0.f) ? m : NEG_SLOPE * m; acc += m * w.x;
        m = a.y + b.y; m = (m > 0.f) ? m : NEG_SLOPE * m; acc += m * w.y;
        m = a.z + b.z; m = (m > 0.f) ? m : NEG_SLOPE * m; acc += m * w.z;
        m = a.w + b.w; m = (m > 0.f) ? m : NEG_SLOPE * m; acc += m * w.w;
    }
    ebuf[idx] = acc;
    atomicMax(&emax[(size_t)d * HH + h], ford(acc));
}

__global__ __launch_bounds__(256)
void edge_exp(const int* __restrict__ ei, int E, int N,
              float* __restrict__ ebuf, const unsigned* __restrict__ emax,
              float* __restrict__ denom) {
    int idx = blockIdx.x * 256 + threadIdx.x;
    int Et = E + N;
    if (idx >= Et * HH) return;
    int et = idx >> 2, h = idx & 3;
    int s, d; edge_ends(ei, E, et, s, d);
    float mx = dford(emax[(size_t)d * HH + h]);
    float ee = __expf(ebuf[idx] - mx);
    ebuf[idx] = ee;                       // reuse buffer in place
    atomicAdd(&denom[(size_t)d * HH + h], ee);
}

__global__ __launch_bounds__(256)
void edge_agg(const int* __restrict__ ei, int E, int N,
              const float* __restrict__ ebuf, const float* __restrict__ denom,
              const float* __restrict__ xl, float* __restrict__ agg) {
    int idx = blockIdx.x * 256 + threadIdx.x;
    int Et = E + N;
    if (idx >= Et * HH) return;
    int et = idx >> 2, h = idx & 3;
    int s, d; edge_ends(ei, E, et, s, d);
    float alpha = ebuf[idx] / (denom[(size_t)d * HH + h] + 1e-16f);
    const float* sp = xl  + (size_t)s * HC + h * CC;
    float*       dp = agg + (size_t)d * HC + h * CC;
    #pragma unroll
    for (int c = 0; c < CC; ++c) atomicAdd(&dp[c], alpha * sp[c]);
}

// h = relu(agg + bias); per-graph max-pool via ordered-uint atomicMax
__global__ __launch_bounds__(256)
void relu_pool(const float* __restrict__ agg, const float* __restrict__ bias,
               const int* __restrict__ batch, unsigned* __restrict__ gord, int N) {
    int i = blockIdx.x * 256 + threadIdx.x;
    if (i >= N * HC) return;
    int n = i >> 7, c = i & 127;
    float hv = fmaxf(agg[i] + bias[c], 0.f);
    atomicMax(&gord[(size_t)batch[n] * HC + c], ford(hv));
}

// Dueling head: one block per graph, 128 threads (one per hidden unit).
__global__ __launch_bounds__(128)
void dueling(const unsigned* __restrict__ gord,
             const float* __restrict__ Wq1, const float* __restrict__ bq1,
             const float* __restrict__ Wq2, const float* __restrict__ bq2,
             const float* __restrict__ Wv1, const float* __restrict__ bv1,
             const float* __restrict__ Wv2, const float* __restrict__ bv2,
             float* __restrict__ out) {
    __shared__ float gl[HC];
    __shared__ float red0[HC], red1[HC], red2[HC];
    int g = blockIdx.x, t = threadIdx.x;
    gl[t] = dford(gord[(size_t)g * HC + t]);
    __syncthreads();
    float hq = bq1[t], hv = bv1[t];
    #pragma unroll 4
    for (int k = 0; k < HC; ++k) {
        float gv = gl[k];
        hq += gv * Wq1[k * HC + t];
        hv += gv * Wv1[k * HC + t];
    }
    hq = fmaxf(hq, 0.f);
    hv = fmaxf(hv, 0.f);
    red0[t] = hq * Wq2[t * 2 + 0];
    red1[t] = hq * Wq2[t * 2 + 1];
    red2[t] = hv * Wv2[t];
    __syncthreads();
    for (int sft = 64; sft > 0; sft >>= 1) {
        if (t < sft) {
            red0[t] += red0[t + sft];
            red1[t] += red1[t + sft];
            red2[t] += red2[t + sft];
        }
        __syncthreads();
    }
    if (t == 0) {
        float q0 = red0[0] + bq2[0];
        float q1 = red1[0] + bq2[1];
        float v  = red2[0] + bv2[0];
        float qm = 0.5f * (q0 + q1);
        out[g * 2 + 0] = q0 - qm + v;
        out[g * 2 + 1] = q1 - qm + v;
    }
}

// ---------------------------------------------------------------------------
extern "C" void kernel_launch(void* const* d_in, const int* in_sizes, int n_in,
                              void* d_out, int out_size, void* d_ws, size_t ws_size,
                              hipStream_t stream) {
    const float* x    = (const float*)d_in[0];
    const float* W_l  = (const float*)d_in[1];
    const float* b_l  = (const float*)d_in[2];
    const float* W_r  = (const float*)d_in[3];
    const float* b_r  = (const float*)d_in[4];
    const float* att  = (const float*)d_in[5];
    const float* bias = (const float*)d_in[6];
    const float* Wq1  = (const float*)d_in[7];
    const float* bq1  = (const float*)d_in[8];
    const float* Wq2  = (const float*)d_in[9];
    const float* bq2  = (const float*)d_in[10];
    const float* Wv1  = (const float*)d_in[11];
    const float* bv1  = (const float*)d_in[12];
    const float* Wv2  = (const float*)d_in[13];
    const float* bv2  = (const float*)d_in[14];
    const int* ei     = (const int*)d_in[15];
    const int* batch  = (const int*)d_in[16];
    float* out        = (float*)d_out;

    const int N  = in_sizes[0] / HC;
    const int E  = in_sizes[15] / 2;
    const int Et = E + N;

    // carve workspace (256B aligned slices)
    char* ws = (char*)d_ws;
    size_t off = 0;
    auto carve = [&](size_t bytes) -> void* {
        void* p = ws + off;
        off = (off + bytes + 255) & ~(size_t)255;
        return p;
    };
    bf16*     x_bf  = (bf16*)carve((size_t)N * HC * sizeof(bf16));
    bf16*     Wl_t  = (bf16*)carve((size_t)HC * HC * sizeof(bf16));
    bf16*     Wr_t  = (bf16*)carve((size_t)HC * HC * sizeof(bf16));
    float*    xl    = (float*)carve((size_t)N * HC * sizeof(float));
    float*    xr    = (float*)carve((size_t)N * HC * sizeof(float));
    float*    ebuf  = (float*)carve((size_t)Et * HH * sizeof(float));
    unsigned* emax  = (unsigned*)carve((size_t)N * HH * sizeof(unsigned));
    float*    denom = (float*)carve((size_t)N * HH * sizeof(float));
    float*    agg   = (float*)carve((size_t)N * HC * sizeof(float));
    unsigned* gord  = (unsigned*)carve((size_t)GG * HC * sizeof(unsigned));
    (void)ws_size; (void)n_in; (void)out_size;

    auto blk = [](int n) { return (unsigned)((n + 255) / 256); };

    // 1) precision conversion + weight transpose
    cvt_bf16<<<blk(N * HC), 256, 0, stream>>>(x, x_bf, N * HC);
    w_transpose_bf16<<<64, 256, 0, stream>>>(W_l, Wl_t);
    w_transpose_bf16<<<64, 256, 0, stream>>>(W_r, Wr_t);

    // 2) node transforms: WMMA GEMMs over full 64-row tiles + scalar tail
    const int mtiles64 = N / 64;
    const int covered  = mtiles64 * 64;
    if (mtiles64 > 0) {
        gemm128_wmma<<<(unsigned)mtiles64, 256, 0, stream>>>(x_bf, Wl_t, b_l, xl, N);
        gemm128_wmma<<<(unsigned)mtiles64, 256, 0, stream>>>(x_bf, Wr_t, b_r, xr, N);
    }
    if (covered < N) {
        unsigned tb = blk((N - covered) * HC);
        gemm_tail<<<tb, 256, 0, stream>>>(x, W_l, b_l, xl, covered, N);
        gemm_tail<<<tb, 256, 0, stream>>>(x, W_r, b_r, xr, covered, N);
    }

    // 3) init reduction buffers
    fill_u32<<<blk(N * HH), 256, 0, stream>>>(emax, ORD_NEG_INF, N * HH);
    fill_f32<<<blk(N * HH), 256, 0, stream>>>(denom, 0.f, N * HH);
    fill_f32<<<blk(N * HC), 256, 0, stream>>>(agg, 0.f, N * HC);
    fill_u32<<<blk(GG * HC), 256, 0, stream>>>(gord, ORD_NEG_INF, GG * HC);

    // 4) edge phase: score -> segment-max, exp -> segment-sum, weighted scatter
    unsigned eblocks = blk(Et * HH);
    edge_score<<<eblocks, 256, 0, stream>>>(xl, xr, att, ei, E, N, ebuf, emax);
    edge_exp<<<eblocks, 256, 0, stream>>>(ei, E, N, ebuf, emax, denom);
    edge_agg<<<eblocks, 256, 0, stream>>>(ei, E, N, ebuf, denom, xl, agg);

    // 5) relu + per-graph max pool
    relu_pool<<<blk(N * HC), 256, 0, stream>>>(agg, bias, batch, gord, N);

    // 6) dueling head
    dueling<<<GG, 128, 0, stream>>>(gord, Wq1, bq1, Wq2, bq2, Wv1, bv1, Wv2, bv2, out);
}
